// TemporalAttention_24541443129668
// MI455X (gfx1250) — compile-verified, matrix operand order checked by reading, not run on hardware
//
#include <hip/hip_runtime.h>
#include <hip/hip_bf16.h>
#include <cmath>

typedef __attribute__((ext_vector_type(16))) __bf16 v16bf;
typedef __attribute__((ext_vector_type(8)))  float  v8f;
typedef __attribute__((ext_vector_type(4)))  unsigned int u32x4;
typedef __attribute__((ext_vector_type(8)))  int i32x8;
typedef __attribute__((ext_vector_type(4)))  int i32x4;

#define BM 128
#define BN 128
#define BK 32
#define LDS_STRIDE 40   // halfs per LDS row: 32 data + 8 pad = 80 bytes (16B aligned)

static constexpr int Bb = 16, Tt = 24, Nn = 325, Dd = 512;
static constexpr int KH = 8, DH = 64;
static constexpr int MROWS = Bb * Tt * Nn;          // 124800 (divisible by 128)

__device__ __forceinline__ unsigned short f2bf(float f) {
    unsigned int u = __float_as_uint(f);
    u += 0x7FFFu + ((u >> 16) & 1u);                // round-to-nearest-even
    return (unsigned short)(u >> 16);
}

__device__ __forceinline__ float gelu_exact(float x) {
    return 0.5f * x * (1.0f + erff(x * 0.70710678118654752440f));
}

// A/B fragment load (16x32 bf16, ISA 7.12.2): lane L -> row rowBase+(L&15);
// halfs [kh*8, kh*8+8) and [16+kh*8, 16+kh*8+8) where kh = L>>4. Two b128 LDS loads.
__device__ __forceinline__ v16bf load_frag(const unsigned short* base, int rowBase,
                                           int stride, int lane) {
    int row = rowBase + (lane & 15);
    int kh  = lane >> 4;
    union { uint4 q[2]; v16bf v; } u;
    const unsigned short* p = base + row * stride + kh * 8;
    u.q[0] = *reinterpret_cast<const uint4*>(p);
    u.q[1] = *reinterpret_cast<const uint4*>(p + 16);
    return u.v;
}

// TDM: copy a rows x 32-half bf16 tile (row stride = strideHalfs in global) into LDS
// with 16B pad after every 64B row (matches LDS_STRIDE=40). ISA 8.3/8.4 D# layout:
// g0: count=1|type=2|lds_addr|global_addr ; g1: data_size=2B, pad 16dw/4dw, dims/strides.
__device__ __forceinline__ void tdm_load_tile(const unsigned short* gsrc,
                                              unsigned short* ldst,
                                              unsigned rows, unsigned strideHalfs) {
    unsigned long long ga = (unsigned long long)(uintptr_t)gsrc;
    unsigned lds = (unsigned)(uintptr_t)ldst;      // low 32 bits of flat = LDS offset
    u32x4 g0;
    g0[0] = 1u;                                    // count=1, load, user mode
    g0[1] = lds;                                   // lds_addr
    g0[2] = (unsigned)ga;                          // global_addr[31:0]
    g0[3] = (unsigned)((ga >> 32) & 0x01FFFFFFu) | (2u << 30);  // addr[56:32] | type=2
    i32x8 g1;
    g1[0] = (1 << 16) | (1 << 20) | (3 << 22) | (3 << 25);
    //        data_size=2B  pad_en   16 DWORDs   4 DWORDs pad
    g1[1] = (int)((strideHalfs & 0xFFFFu) << 16);               // tensor_dim0[15:0]
    g1[2] = (int)((strideHalfs >> 16) & 0xFFFFu) | (int)((rows & 0xFFFFu) << 16);
    g1[3] = (int)((rows >> 16) & 0xFFFFu) | (32 << 16);         // tile_dim0 = 32
    g1[4] = (int)rows;                                          // tile_dim1, tile_dim2=0
    g1[5] = (int)strideHalfs;                                   // tensor_dim0_stride lo
    g1[6] = 0;
    g1[7] = 0;
    i32x4 z4 = {0, 0, 0, 0};
    i32x8 z8 = {0, 0, 0, 0, 0, 0, 0, 0};
    __builtin_amdgcn_tensor_load_to_lds(g0, g1, z4, z4, z8, 0);
}

// ---------------------------------------------------------------- weights prep
__global__ __launch_bounds__(256) void prep_weights(
    const float* __restrict__ Wqkv, const float* __restrict__ W1,
    const float* __restrict__ W2,
    unsigned short* __restrict__ WqkvT, unsigned short* __restrict__ W1T,
    unsigned short* __restrict__ W2T)
{
    int idx = blockIdx.x * 256 + threadIdx.x;
    if (idx < 1024 * 1024) {
        int k = idx >> 10, n = idx & 1023;
        WqkvT[n * 1024 + k] = f2bf(Wqkv[idx]);
    }
    if (idx < 512 * 512) {
        int k = idx >> 9, n = idx & 511;
        W1T[n * 512 + k] = f2bf(W1[idx]);
        W2T[n * 512 + k] = f2bf(W2[idx]);
    }
}

// ---------------------------------------------------------------- QKV GEMM
// out[r][c] = gelu(concat(x,STE)[r] . Wqkv[:,c] + b[c]),  M=124800 K=1024 N=1024
__global__ __launch_bounds__(256) void qkv_gemm(
    const float* __restrict__ x, const float* __restrict__ ste,
    const unsigned short* __restrict__ WT, const float* __restrict__ bias,
    unsigned short* __restrict__ out)
{
    const int Kd = 1024, NK = Kd / BK;
    __shared__ unsigned short As[2][BM * LDS_STRIDE];
    __shared__ unsigned short Bs[2][BN * LDS_STRIDE];

    int tid = threadIdx.x, lane = tid & 31, wave = tid >> 5;
    int wm = wave >> 1, wn = wave & 1;
    int bm = blockIdx.x >> 3, bn = blockIdx.x & 7;
    int row0 = bm * BM, col0 = bn * BN;

    v8f acc[2][4];
    #pragma unroll
    for (int i = 0; i < 2; i++)
        #pragma unroll
        for (int j = 0; j < 4; j++) acc[i][j] = {};

    float4 areg[4];
    auto loadA = [&](int k0) {
        #pragma unroll
        for (int i = 0; i < 4; i++) {
            int f4 = tid + 256 * i;
            int r = f4 >> 3, c4 = f4 & 7;
            int gc = k0 + c4 * 4;
            const float* src = (gc < 512) ? (x   + (size_t)(row0 + r) * 512 + gc)
                                          : (ste + (size_t)(row0 + r) * 512 + (gc - 512));
            areg[i] = *reinterpret_cast<const float4*>(src);
        }
    };
    auto storeA = [&](unsigned short* Abuf) {
        #pragma unroll
        for (int i = 0; i < 4; i++) {
            int f4 = tid + 256 * i;
            int r = f4 >> 3, c4 = f4 & 7;
            unsigned short* dst = &Abuf[r * LDS_STRIDE + c4 * 4];
            dst[0] = f2bf(areg[i].x); dst[1] = f2bf(areg[i].y);
            dst[2] = f2bf(areg[i].z); dst[3] = f2bf(areg[i].w);
        }
    };
    auto compute = [&](const unsigned short* Abuf, const unsigned short* Bbuf) {
        v16bf af[2], bfr[4];
        #pragma unroll
        for (int mi = 0; mi < 2; mi++) af[mi] = load_frag(Abuf, wm * 32 + mi * 16, LDS_STRIDE, lane);
        #pragma unroll
        for (int ni = 0; ni < 4; ni++) bfr[ni] = load_frag(Bbuf, wn * 64 + ni * 16, LDS_STRIDE, lane);
        #pragma unroll
        for (int mi = 0; mi < 2; mi++)
            #pragma unroll
            for (int ni = 0; ni < 4; ni++)
                acc[mi][ni] = __builtin_amdgcn_wmma_f32_16x16x32_bf16(
                    false, af[mi], false, bfr[ni], (short)0, acc[mi][ni], false, false);
    };

    // prologue: stage tile 0
    if (wave == 0) tdm_load_tile(WT + (size_t)col0 * Kd, Bs[0], BN, Kd);
    loadA(0);
    storeA(As[0]);
    if (wave == 0) __builtin_amdgcn_s_wait_tensorcnt(0);
    __syncthreads();

    for (int k = 0; k < NK; k++) {
        int cur = k & 1, nxt = cur ^ 1;
        if (k + 1 < NK) {
            if (wave == 0)
                tdm_load_tile(WT + (size_t)col0 * Kd + (k + 1) * BK, Bs[nxt], BN, Kd);
            loadA((k + 1) * BK);            // global loads in flight during WMMA
        }
        compute(As[cur], Bs[cur]);
        if (k + 1 < NK) {
            storeA(As[nxt]);
            if (wave == 0) __builtin_amdgcn_s_wait_tensorcnt(0);
        }
        __syncthreads();
    }

    #pragma unroll
    for (int mi = 0; mi < 2; mi++)
        #pragma unroll
        for (int ni = 0; ni < 4; ni++) {
            int col = col0 + wn * 64 + ni * 16 + (lane & 15);
            float bv = bias[col];
            int rbase = row0 + wm * 32 + mi * 16 + (lane >> 4) * 8;
            #pragma unroll
            for (int i = 0; i < 8; i++)
                out[(size_t)(rbase + i) * 1024 + col] = f2bf(gelu_exact(acc[mi][ni][i] + bv));
        }
}

// ---------------------------------------------------------------- attention
// One head per wave; 4 waves per block. S=qk.qk^T/8, softmax over s<24, out=att.v + x
__global__ __launch_bounds__(128) void attn_kernel(
    const unsigned short* __restrict__ qkv, const float* __restrict__ x,
    float* __restrict__ out_attn)
{
    __shared__ unsigned short qs[4][32 * 64];
    __shared__ unsigned short vs[4][64 * 32];       // v transposed: [d][t]
    __shared__ float          Ss[4][32 * 32];
    __shared__ unsigned short as_[4][32 * 32];

    int lane = threadIdx.x & 31, wave = threadIdx.x >> 5;
    int h = blockIdx.x * 4 + wave;                  // 41600 heads total
    int b = h / (Nn * KH);
    int rem = h % (Nn * KH);
    int n = rem / KH, head = rem % KH;

    unsigned short* q  = qs[wave];
    unsigned short* vT = vs[wave];
    float*          S  = Ss[wave];
    unsigned short* att = as_[wave];

    for (int i = lane; i < 64 * 32; i += 32) vT[i] = 0;
    for (int i = lane; i < 8 * 64; i += 32) q[24 * 64 + i] = 0;

    for (int idx = lane; idx < 24 * 8; idx += 32) {
        int t = idx >> 3, c = idx & 7;
        size_t r = (size_t)(b * Tt + t) * Nn + n;
        uint4 v = *reinterpret_cast<const uint4*>(qkv + r * 1024 + head * 128 + c * 8);
        *reinterpret_cast<uint4*>(&q[t * 64 + c * 8]) = v;
    }
    for (int idx = lane; idx < 24 * 8; idx += 32) {
        int t = idx >> 3, c = idx & 7;
        size_t r = (size_t)(b * Tt + t) * Nn + n;
        union { uint4 u; unsigned short s[8]; } w;
        w.u = *reinterpret_cast<const uint4*>(qkv + r * 1024 + head * 128 + 64 + c * 8);
        #pragma unroll
        for (int j = 0; j < 8; j++) vT[(c * 8 + j) * 32 + t] = w.s[j];
    }
    __syncthreads();

    v8f sa[2][2];
    #pragma unroll
    for (int mi = 0; mi < 2; mi++)
        #pragma unroll
        for (int ni = 0; ni < 2; ni++) sa[mi][ni] = {};
    #pragma unroll
    for (int kk = 0; kk < 64; kk += 32) {
        v16bf a[2], bb[2];
        #pragma unroll
        for (int mi = 0; mi < 2; mi++) a[mi]  = load_frag(q + kk, mi * 16, 64, lane);
        #pragma unroll
        for (int ni = 0; ni < 2; ni++) bb[ni] = load_frag(q + kk, ni * 16, 64, lane);
        #pragma unroll
        for (int mi = 0; mi < 2; mi++)
            #pragma unroll
            for (int ni = 0; ni < 2; ni++)
                sa[mi][ni] = __builtin_amdgcn_wmma_f32_16x16x32_bf16(
                    false, a[mi], false, bb[ni], (short)0, sa[mi][ni], false, false);
    }
    #pragma unroll
    for (int mi = 0; mi < 2; mi++)
        #pragma unroll
        for (int ni = 0; ni < 2; ni++) {
            int col = ni * 16 + (lane & 15);
            int rb = mi * 16 + (lane >> 4) * 8;
            #pragma unroll
            for (int i = 0; i < 8; i++)
                S[(rb + i) * 32 + col] = sa[mi][ni][i] * 0.125f;  // 1/sqrt(64)
        }
    __syncthreads();

    if (lane < 24) {
        float m = -1e30f;
        for (int s = 0; s < 24; s++) m = fmaxf(m, S[lane * 32 + s]);
        float e[24], sum = 0.f;
        for (int s = 0; s < 24; s++) { e[s] = __expf(S[lane * 32 + s] - m); sum += e[s]; }
        float inv = 1.0f / sum;
        for (int s = 0; s < 24; s++) att[lane * 32 + s] = f2bf(e[s] * inv);
        for (int s = 24; s < 32; s++) att[lane * 32 + s] = 0;
    } else {
        for (int s = 0; s < 32; s++) att[lane * 32 + s] = 0;
    }
    __syncthreads();

    v8f oa[2][4];
    #pragma unroll
    for (int mi = 0; mi < 2; mi++)
        #pragma unroll
        for (int ni = 0; ni < 4; ni++) oa[mi][ni] = {};
    {
        v16bf a[2], bb[4];
        #pragma unroll
        for (int mi = 0; mi < 2; mi++) a[mi]  = load_frag(att, mi * 16, 32, lane);
        #pragma unroll
        for (int ni = 0; ni < 4; ni++) bb[ni] = load_frag(vT, ni * 16, 32, lane);
        #pragma unroll
        for (int mi = 0; mi < 2; mi++)
            #pragma unroll
            for (int ni = 0; ni < 4; ni++)
                oa[mi][ni] = __builtin_amdgcn_wmma_f32_16x16x32_bf16(
                    false, a[mi], false, bb[ni], (short)0, oa[mi][ni], false, false);
    }
    #pragma unroll
    for (int mi = 0; mi < 2; mi++)
        #pragma unroll
        for (int ni = 0; ni < 4; ni++) {
            int d = ni * 16 + (lane & 15);
            int tb = mi * 16 + (lane >> 4) * 8;
            #pragma unroll
            for (int i = 0; i < 8; i++) {
                int t = tb + i;
                if (t < 24) {
                    size_t r = (size_t)(b * Tt + t) * Nn + n;
                    int gc = head * 64 + d;
                    out_attn[r * 512 + gc] = oa[mi][ni][i] + x[r * 512 + gc];
                }
            }
        }
}

// ---------------------------------------------------------------- FFN1: hidden = gelu(A@W1+b1)
__global__ __launch_bounds__(256) void ffn1_gemm(
    const float* __restrict__ A, const unsigned short* __restrict__ WT,
    const float* __restrict__ bias, unsigned short* __restrict__ out)
{
    const int Kd = 512, NK = Kd / BK;
    __shared__ unsigned short As[2][BM * LDS_STRIDE];
    __shared__ unsigned short Bs[2][BN * LDS_STRIDE];
    int tid = threadIdx.x, lane = tid & 31, wave = tid >> 5;
    int wm = wave >> 1, wn = wave & 1;
    int bm = blockIdx.x >> 2, bn = blockIdx.x & 3;
    int row0 = bm * BM, col0 = bn * BN;

    v8f acc[2][4];
    #pragma unroll
    for (int i = 0; i < 2; i++)
        #pragma unroll
        for (int j = 0; j < 4; j++) acc[i][j] = {};

    float4 areg[4];
    auto loadA = [&](int k0) {
        #pragma unroll
        for (int i = 0; i < 4; i++) {
            int f4 = tid + 256 * i;
            int r = f4 >> 3, c4 = f4 & 7;
            areg[i] = *reinterpret_cast<const float4*>(
                A + (size_t)(row0 + r) * 512 + k0 + c4 * 4);
        }
    };
    auto storeA = [&](unsigned short* Abuf) {
        #pragma unroll
        for (int i = 0; i < 4; i++) {
            int f4 = tid + 256 * i;
            int r = f4 >> 3, c4 = f4 & 7;
            unsigned short* dst = &Abuf[r * LDS_STRIDE + c4 * 4];
            dst[0] = f2bf(areg[i].x); dst[1] = f2bf(areg[i].y);
            dst[2] = f2bf(areg[i].z); dst[3] = f2bf(areg[i].w);
        }
    };
    auto compute = [&](const unsigned short* Abuf, const unsigned short* Bbuf) {
        v16bf af[2], bfr[4];
        #pragma unroll
        for (int mi = 0; mi < 2; mi++) af[mi] = load_frag(Abuf, wm * 32 + mi * 16, LDS_STRIDE, lane);
        #pragma unroll
        for (int ni = 0; ni < 4; ni++) bfr[ni] = load_frag(Bbuf, wn * 64 + ni * 16, LDS_STRIDE, lane);
        #pragma unroll
        for (int mi = 0; mi < 2; mi++)
            #pragma unroll
            for (int ni = 0; ni < 4; ni++)
                acc[mi][ni] = __builtin_amdgcn_wmma_f32_16x16x32_bf16(
                    false, af[mi], false, bfr[ni], (short)0, acc[mi][ni], false, false);
    };

    if (wave == 0) tdm_load_tile(WT + (size_t)col0 * Kd, Bs[0], BN, Kd);
    loadA(0);
    storeA(As[0]);
    if (wave == 0) __builtin_amdgcn_s_wait_tensorcnt(0);
    __syncthreads();

    for (int k = 0; k < NK; k++) {
        int cur = k & 1, nxt = cur ^ 1;
        if (k + 1 < NK) {
            if (wave == 0)
                tdm_load_tile(WT + (size_t)col0 * Kd + (k + 1) * BK, Bs[nxt], BN, Kd);
            loadA((k + 1) * BK);
        }
        compute(As[cur], Bs[cur]);
        if (k + 1 < NK) {
            storeA(As[nxt]);
            if (wave == 0) __builtin_amdgcn_s_wait_tensorcnt(0);
        }
        __syncthreads();
    }

    #pragma unroll
    for (int mi = 0; mi < 2; mi++)
        #pragma unroll
        for (int ni = 0; ni < 4; ni++) {
            int col = col0 + wn * 64 + ni * 16 + (lane & 15);
            float bv = bias[col];
            int rbase = row0 + wm * 32 + mi * 16 + (lane >> 4) * 8;
            #pragma unroll
            for (int i = 0; i < 8; i++)
                out[(size_t)(rbase + i) * 512 + col] = f2bf(gelu_exact(acc[mi][ni][i] + bv));
        }
}

// ---------------------------------------------------------------- FFN2: out = hidden@W2+b2+resid
// Both A (bf16 hidden) and B tiles stream in via TDM; waves only do WMMA + epilogue.
__global__ __launch_bounds__(256) void ffn2_gemm(
    const unsigned short* __restrict__ A, const unsigned short* __restrict__ WT,
    const float* __restrict__ bias, const float* __restrict__ resid,
    float* __restrict__ out)
{
    const int Kd = 512, NK = Kd / BK;
    __shared__ unsigned short As[2][BM * LDS_STRIDE];
    __shared__ unsigned short Bs[2][BN * LDS_STRIDE];
    int tid = threadIdx.x, lane = tid & 31, wave = tid >> 5;
    int wm = wave >> 1, wn = wave & 1;
    int bm = blockIdx.x >> 2, bn = blockIdx.x & 3;
    int row0 = bm * BM, col0 = bn * BN;

    v8f acc[2][4];
    #pragma unroll
    for (int i = 0; i < 2; i++)
        #pragma unroll
        for (int j = 0; j < 4; j++) acc[i][j] = {};

    auto issue_tiles = [&](int k0, int buf) {
        tdm_load_tile(A  + (size_t)row0 * Kd + k0, As[buf], BM, Kd);
        tdm_load_tile(WT + (size_t)col0 * Kd + k0, Bs[buf], BN, Kd);
    };
    auto compute = [&](const unsigned short* Abuf, const unsigned short* Bbuf) {
        v16bf af[2], bfr[4];
        #pragma unroll
        for (int mi = 0; mi < 2; mi++) af[mi] = load_frag(Abuf, wm * 32 + mi * 16, LDS_STRIDE, lane);
        #pragma unroll
        for (int ni = 0; ni < 4; ni++) bfr[ni] = load_frag(Bbuf, wn * 64 + ni * 16, LDS_STRIDE, lane);
        #pragma unroll
        for (int mi = 0; mi < 2; mi++)
            #pragma unroll
            for (int ni = 0; ni < 4; ni++)
                acc[mi][ni] = __builtin_amdgcn_wmma_f32_16x16x32_bf16(
                    false, af[mi], false, bfr[ni], (short)0, acc[mi][ni], false, false);
    };

    if (wave == 0) {
        issue_tiles(0, 0);
        __builtin_amdgcn_s_wait_tensorcnt(0);
    }
    __syncthreads();

    for (int k = 0; k < NK; k++) {
        int cur = k & 1, nxt = cur ^ 1;
        if (wave == 0 && k + 1 < NK) issue_tiles((k + 1) * BK, nxt);
        compute(As[cur], Bs[cur]);
        if (wave == 0 && k + 1 < NK) __builtin_amdgcn_s_wait_tensorcnt(0);
        __syncthreads();
    }

    #pragma unroll
    for (int mi = 0; mi < 2; mi++)
        #pragma unroll
        for (int ni = 0; ni < 4; ni++) {
            int col = col0 + wn * 64 + ni * 16 + (lane & 15);
            float bv = bias[col];
            int rbase = row0 + wm * 32 + mi * 16 + (lane >> 4) * 8;
            #pragma unroll
            for (int i = 0; i < 8; i++) {
                size_t o = (size_t)(rbase + i) * 512 + col;
                out[o] = acc[mi][ni][i] + bv + resid[o];
            }
        }
}

// ---------------------------------------------------------------- launcher
extern "C" void kernel_launch(void* const* d_in, const int* in_sizes, int n_in,
                              void* d_out, int out_size, void* d_ws, size_t ws_size,
                              hipStream_t stream) {
    const float* x    = (const float*)d_in[0];
    const float* STE  = (const float*)d_in[1];
    const float* Wqkv = (const float*)d_in[2];
    const float* bqkv = (const float*)d_in[3];
    const float* W1   = (const float*)d_in[4];
    const float* b1   = (const float*)d_in[5];
    const float* W2   = (const float*)d_in[6];
    const float* b2   = (const float*)d_in[7];
    float* out = (float*)d_out;

    char* ws = (char*)d_ws;
    size_t off = 0;
    auto take = [&](size_t bytes) -> char* {
        off = (off + 255) & ~(size_t)255;
        char* p = ws + off;
        off += bytes;
        return p;
    };
    unsigned short* WqkvT   = (unsigned short*)take((size_t)1024 * 1024 * 2);
    unsigned short* W1T     = (unsigned short*)take((size_t)512 * 512 * 2);
    unsigned short* W2T     = (unsigned short*)take((size_t)512 * 512 * 2);
    unsigned short* qkv_ws  = (unsigned short*)take((size_t)MROWS * 1024 * 2);
    float*          attn_ws = (float*)take((size_t)MROWS * 512 * 4);
    unsigned short* hid_ws  = (unsigned short*)take((size_t)MROWS * 512 * 2);
    (void)ws_size; (void)in_sizes; (void)n_in; (void)out_size;

    prep_weights<<<(1024 * 1024) / 256, 256, 0, stream>>>(Wqkv, W1, W2, WqkvT, W1T, W2T);

    qkv_gemm<<<975 * 8, 256, 0, stream>>>(x, STE, WqkvT, bqkv, qkv_ws);

    attn_kernel<<<41600 / 4, 128, 0, stream>>>(qkv_ws, x, attn_ws);

    ffn1_gemm<<<975 * 4, 256, 0, stream>>>(attn_ws, W1T, b1, hid_ws);
    ffn2_gemm<<<975 * 4, 256, 0, stream>>>(hid_ws, W2T, b2, attn_ws, out);
}
